// FusedLoRAAdapter_16561393893679
// MI455X (gfx1250) — matrix operand order repaired and checked by reading, not compiled, not run on hardware
//
#include <hip/hip_runtime.h>
#include <stdint.h>

// ---------------------------------------------------------------------------
// Fused LoRA adapter for MI455X (gfx1250):
//   y = x @ (W + (alpha/r) * B@A)^T + bias
// Round 2: fold LoRA into W (bf16), pre-convert x to bf16, then a bf16 WMMA
// GEMM whose K-pipeline uses GLOBAL_LOAD_ASYNC_TO_LDS_B128 (ASYNCcnt) for
// global->LDS staging. No in-loop cvt, no register staging -> no spills.
// ---------------------------------------------------------------------------

typedef __attribute__((ext_vector_type(16))) __bf16 v16bf;
typedef __attribute__((ext_vector_type(8)))  float  v8f;

#define D_K 4096          // inner dim (D_in)
#define D_N 4096          // output dim (D_out)
#define BM  128
#define BN  128
#define BK  32
#define LDSP 40           // padded LDS row pitch (bf16 elems) = 80B -> bank-conflict-free
#define BUFB (BM * LDSP * 2)   // bytes per LDS buffer (10240)

__device__ __forceinline__ uint16_t f2bf(float f) {
  uint32_t u = __float_as_uint(f);
  u += 0x7FFFu + ((u >> 16) & 1u);      // round-to-nearest-even
  return (uint16_t)(u >> 16);
}

// ---------------------------------------------------------------------------
// Prep 1: Wp[n,k] = bf16( W[n,k] + scale * sum_r B[n,r]*A[r,k] )
// ---------------------------------------------------------------------------
__global__ __launch_bounds__(256)
void lora_prep_kernel(const float* __restrict__ W,
                      const float* __restrict__ A,
                      const float* __restrict__ Bm,
                      uint16_t* __restrict__ Wp,
                      float scale, int R) {
  const int n  = blockIdx.x;
  const int k0 = threadIdx.x * 16;

  float hb[16];
#pragma unroll
  for (int r = 0; r < 16; ++r)
    hb[r] = (r < R) ? Bm[(size_t)n * R + r] * scale : 0.0f;

  float acc[16];
  const float4* wv = (const float4*)(W + (size_t)n * D_K + k0);
#pragma unroll
  for (int i = 0; i < 4; ++i) {
    float4 t = wv[i];
    acc[4*i+0] = t.x; acc[4*i+1] = t.y; acc[4*i+2] = t.z; acc[4*i+3] = t.w;
  }
#pragma unroll
  for (int r = 0; r < 16; ++r) {
    const float4* av = (const float4*)(A + (size_t)r * D_K + k0);
#pragma unroll
    for (int i = 0; i < 4; ++i) {
      float4 t = av[i];
      acc[4*i+0] = fmaf(hb[r], t.x, acc[4*i+0]);
      acc[4*i+1] = fmaf(hb[r], t.y, acc[4*i+1]);
      acc[4*i+2] = fmaf(hb[r], t.z, acc[4*i+2]);
      acc[4*i+3] = fmaf(hb[r], t.w, acc[4*i+3]);
    }
  }

  uint16_t o[16];
#pragma unroll
  for (int i = 0; i < 16; ++i) o[i] = f2bf(acc[i]);
  uint4* dst = (uint4*)(Wp + (size_t)n * D_K + k0);
  dst[0] = ((const uint4*)o)[0];
  dst[1] = ((const uint4*)o)[1];
}

// ---------------------------------------------------------------------------
// Prep 2: Xb = bf16(x), streaming convert, 8 elems/thread
// ---------------------------------------------------------------------------
__global__ __launch_bounds__(256)
void x_to_bf16_kernel(const float* __restrict__ X, uint16_t* __restrict__ Xb) {
  const size_t i = ((size_t)blockIdx.x * 256 + threadIdx.x) * 8;
  float4 a = *(const float4*)(X + i);
  float4 b = *(const float4*)(X + i + 4);
  uint16_t o[8] = { f2bf(a.x), f2bf(a.y), f2bf(a.z), f2bf(a.w),
                    f2bf(b.x), f2bf(b.y), f2bf(b.z), f2bf(b.w) };
  *(uint4*)(Xb + i) = *(const uint4*)o;
}

// ---------------------------------------------------------------------------
// GEMM: Y[m,n] = sum_k Xb[m,k] * Wp[n,k] + bias[n]
// 256 threads = 8 wave32; block tile 128x128; K-step 32.
// Each wave: 64(M) x 32(N) = 4x2 grid of v_wmma_f32_16x16x32_bf16.
// Global->LDS staging via GLOBAL_LOAD_ASYNC_TO_LDS_B128, double-buffered.
// ---------------------------------------------------------------------------
__global__ __launch_bounds__(256)
void lora_gemm_kernel(const uint16_t* __restrict__ Xb,
                      const uint16_t* __restrict__ Wp,
                      const float* __restrict__ bias,
                      float* __restrict__ Y,
                      int M) {
  __shared__ __align__(16) uint16_t sA[2][BM * LDSP];
  __shared__ __align__(16) uint16_t sB[2][BN * LDSP];

  const int tid  = threadIdx.x;
  const int lane = tid & 31;
  const int l15  = lane & 15;
  const int lh   = lane >> 4;       // 0/1 half-wave
  const int wave = tid >> 5;        // 0..7
  const int wm   = wave >> 2;       // 0..1  -> 64-row slab
  const int wn   = wave & 3;        // 0..3  -> 32-col slab

  const int blockN = blockIdx.x * BN;
  const int blockM = blockIdx.y * BM;

  // staging coords: each thread copies 16 bf16 of one row per tile (2 x b128)
  const int srow = tid >> 1;            // 0..127
  const int scol = (tid & 1) * 16;      // 0 or 16

  const uint16_t* xg = Xb + (size_t)(blockM + srow) * D_K + scol;
  const uint16_t* wg = Wp + (size_t)(blockN + srow) * D_K + scol;

  // LDS byte offsets for async destinations (flat addr low 32 bits = LDS offset)
  const uint32_t aoff = (uint32_t)(uintptr_t)(&sA[0][srow * LDSP + scol]);
  const uint32_t boff = (uint32_t)(uintptr_t)(&sB[0][srow * LDSP + scol]);

  v8f acc[4][2];
#pragma unroll
  for (int mt = 0; mt < 4; ++mt)
#pragma unroll
    for (int nt = 0; nt < 2; ++nt)
      acc[mt][nt] = (v8f)(0.0f);

  const int steps = D_K / BK;   // 128

  // Issue 4 async b128 global->LDS copies for K-step kt into buffer buf.
  auto issue_async = [&](int kt, int buf) {
    const uint32_t bb = (uint32_t)buf * BUFB;
    const uint64_t xa = (uint64_t)(uintptr_t)(xg + (size_t)kt * BK);
    const uint64_t wa = (uint64_t)(uintptr_t)(wg + (size_t)kt * BK);
    asm volatile("global_load_async_to_lds_b128 %0, %1, off"
                 :: "v"(aoff + bb),      "v"(xa)      : "memory");
    asm volatile("global_load_async_to_lds_b128 %0, %1, off"
                 :: "v"(aoff + bb + 16), "v"(xa + 16) : "memory");
    asm volatile("global_load_async_to_lds_b128 %0, %1, off"
                 :: "v"(boff + bb),      "v"(wa)      : "memory");
    asm volatile("global_load_async_to_lds_b128 %0, %1, off"
                 :: "v"(boff + bb + 16), "v"(wa + 16) : "memory");
  };

  union Frag { v16bf v; uint4 u[2]; };

  auto compute = [&](int buf) {
    Frag a[4], b[2];
    // A fragments (16x32 bf16): lane l15 = row M; halves hold K 0-7/16-23 or 8-15/24-31
#pragma unroll
    for (int mt = 0; mt < 4; ++mt) {
      const uint16_t* p = &sA[buf][(wm * 64 + mt * 16 + l15) * LDSP + lh * 8];
      a[mt].u[0] = *(const uint4*)p;
      a[mt].u[1] = *(const uint4*)(p + 16);
    }
    // B fragments (32x16 bf16): lane l15 = col N; halves hold K 0-7/16-23 or 8-15/24-31
#pragma unroll
    for (int nt = 0; nt < 2; ++nt) {
      const uint16_t* p = &sB[buf][(wn * 32 + nt * 16 + l15) * LDSP + lh * 8];
      b[nt].u[0] = *(const uint4*)p;
      b[nt].u[1] = *(const uint4*)(p + 16);
    }
#pragma unroll
    for (int mt = 0; mt < 4; ++mt)
#pragma unroll
      for (int nt = 0; nt < 2; ++nt)
        acc[mt][nt] = __builtin_amdgcn_wmma_f32_16x16x32_bf16(
            false, a[mt].v, false, b[nt].v,
            (short)0, acc[mt][nt], false, false);
  };

  // Pipeline:
  //   issue(0 -> buf0)
  //   for kt: wait asynccnt 0; barrier; issue(kt+1 -> cur^1); compute(cur)
  // The barrier at the top of kt also guarantees every wave finished READING
  // buf cur^1 (its last read was the compute of kt-1), so the new async
  // writes into it are race-free.  One barrier per K-step.
  issue_async(0, 0);

  for (int kt = 0; kt < steps; ++kt) {
    const int cur = kt & 1;
    asm volatile("s_wait_asynccnt 0x0" ::: "memory");
    __syncthreads();
    if (kt + 1 < steps) {
      issue_async(kt + 1, cur ^ 1);
      if (kt + 2 < steps) {
        __builtin_prefetch(xg + (size_t)(kt + 2) * BK, 0, 3);  // global_prefetch_b8
        __builtin_prefetch(wg + (size_t)(kt + 2) * BK, 0, 3);
      }
    }
    compute(cur);
  }

  // epilogue: C/D layout -> VGPR v holds rows (v, v+8), lanes 0-15 are N
  float bv[2];
#pragma unroll
  for (int nt = 0; nt < 2; ++nt)
    bv[nt] = bias[blockN + wn * 32 + nt * 16 + l15];

#pragma unroll
  for (int mt = 0; mt < 4; ++mt) {
#pragma unroll
    for (int v = 0; v < 8; ++v) {
      const int row = blockM + wm * 64 + mt * 16 + lh * 8 + v;
      float* yrow = Y + (size_t)row * D_N + blockN + wn * 32;
#pragma unroll
      for (int nt = 0; nt < 2; ++nt)
        yrow[nt * 16 + l15] = acc[mt][nt][v] + bv[nt];
    }
  }
}

// ---------------------------------------------------------------------------
extern "C" void kernel_launch(void* const* d_in, const int* in_sizes, int n_in,
                              void* d_out, int out_size, void* d_ws, size_t ws_size,
                              hipStream_t stream) {
  const float* x    = (const float*)d_in[0];   // [B,S,D_in] f32
  const float* W    = (const float*)d_in[1];   // [D_out,D_in] f32
  const float* A    = (const float*)d_in[2];   // [R,D_in] f32
  const float* Bm   = (const float*)d_in[3];   // [D_out,R] f32
  const float* bias = (const float*)d_in[4];   // [D_out] f32
  float* out = (float*)d_out;

  const int M = in_sizes[0] / D_K;             // B*S = 8192
  const int R = in_sizes[2] / D_K;             // 16
  const float scale = 16.0f / (float)R;        // LORA_ALPHA / rank

  uint16_t* Wp = (uint16_t*)d_ws;                                  // 32 MB
  uint16_t* Xb = (uint16_t*)((char*)d_ws + (size_t)D_N * D_K * 2); // 64 MB

  lora_prep_kernel<<<D_N, 256, 0, stream>>>(W, A, Bm, Wp, scale, R);

  const int xblocks = (M * D_K) / (256 * 8);
  x_to_bf16_kernel<<<xblocks, 256, 0, stream>>>(x, Xb);

  dim3 grid(D_N / BN, M / BM);
  lora_gemm_kernel<<<grid, 256, 0, stream>>>(Xb, Wp, bias, out, M);
}
